// MultiModalAttention_17171279249428
// MI455X (gfx1250) — matrix operand order repaired and checked by reading, not compiled, not run on hardware
//
#include <hip/hip_runtime.h>
#include <stdint.h>

typedef __attribute__((ext_vector_type(16))) __bf16 v16bf;
typedef __attribute__((ext_vector_type(8)))  float  v8f;

#if defined(__AMDGCN__) && __has_builtin(__builtin_amdgcn_tensor_load_to_lds) && __has_builtin(__builtin_amdgcn_s_wait_tensorcnt)
#define USE_TDM 1
#else
#define USE_TDM 0
#endif

#define DMODEL 1024
#define NHEAD  16
#define DKH    64
#define BATCH  4
#define SEQ    1024
#define MROWS  (BATCH * SEQ)   // 4096

__device__ __forceinline__ __bf16 f2bf(float f) {
  union { float f; uint32_t u; } v; v.f = f;
  uint32_t r = v.u + 0x7FFFu + ((v.u >> 16) & 1u);  // round-to-nearest-even
  union { uint16_t u; __bf16 b; } o; o.u = (uint16_t)(r >> 16);
  return o.b;
}

// pack two fp32 -> one dword of 2 bf16 (lo=a, hi=b)
__device__ __forceinline__ uint32_t bfpack(float a, float b) {
#if __has_builtin(__builtin_amdgcn_cvt_pk_bf16_f32)
  typedef __attribute__((ext_vector_type(2))) __bf16 v2bf;
  union { v2bf v; uint32_t u; } c;
  c.v = __builtin_amdgcn_cvt_pk_bf16_f32(a, b);
  return c.u;
#else
  union { float f; uint32_t u; } x, y; x.f = a; y.f = b;
  uint32_t ra = (x.u + 0x7FFFu + ((x.u >> 16) & 1u)) >> 16;
  uint32_t rb = (y.u + 0x7FFFu + ((y.u >> 16) & 1u)) & 0xFFFF0000u;
  return rb | ra;
#endif
}

#if USE_TDM
typedef uint32_t u32x4 __attribute__((ext_vector_type(4)));
typedef int      i32x8 __attribute__((ext_vector_type(8)));
typedef int      i32x4 __attribute__((ext_vector_type(4)));

// TDM: DMA a 64x64 bf16 tile (row stride 64 elems in memory) into LDS,
// padding +1 dword after every 32 dwords -> LDS row stride 66 bf16.
__device__ __forceinline__ void tdm_load_tile64x64(const __bf16* gsrc, uint32_t lds_off) {
  uint64_t ga = (uint64_t)(uintptr_t)gsrc;
  u32x4 g0;
  g0.x = 0x1u;                                  // count=1 (valid descriptor)
  g0.y = lds_off;                               // LDS byte offset
  g0.z = (uint32_t)ga;                          // global_addr[31:0]
  g0.w = (uint32_t)(ga >> 32) | 0x80000000u;    // global_addr[56:32] | type=2
  i32x8 g1;
  g1[0] = (int)((1u << 16)                      // data_size = 2 bytes
              | (1u << 20)                      // pad_enable
              | (4u << 22));                    // pad_interval = 32 dwords
                                                // pad_amount = 0 (=1 dword)
  g1[1] = (int)(64u << 16);                     // tensor_dim0 = 64
  g1[2] = (int)(64u << 16);                     // tensor_dim1 = 64
  g1[3] = (int)(64u << 16);                     // tile_dim0   = 64
  g1[4] = 64;                                   // tile_dim1   = 64
  g1[5] = 64;                                   // tensor_dim0_stride = 64
  g1[6] = 0; g1[7] = 0;
  i32x4 g2 = {0, 0, 0, 0}, g3 = {0, 0, 0, 0};
  i32x8 g4 = {0, 0, 0, 0, 0, 0, 0, 0};          // 6-arg toolchain: extra group
  __builtin_amdgcn_tensor_load_to_lds(g0, g1, g2, g3, g4, 0);
}
#endif

// ---------------------------------------------------------------------------
// Tiled GEMM, bf16 WMMA:  out = A[4096x1024] @ W[1024x1024] + bias
// ABF16 (template): A is bf16 (workspace) vs fp32 (inputs, converted in-stage)
// MODE  (template): 0 = fp32 row-major out; 1 = bf16 split-head [B,H,L,DKH]
// Block = 64(M) x 128(N) C-tile, 8 waves (2x4), wave = 32x32 (4 WMMA/K-step),
// K stepped by 32, double-buffered LDS staging.
// ---------------------------------------------------------------------------
template <int ABF16, int MODE>
__global__ __launch_bounds__(256)
void gemm_bias_kernel(const void* __restrict__ Aptr,
                      const float* __restrict__ W,
                      const float* __restrict__ bias,
                      void* __restrict__ outp)
{
  __shared__ __bf16 As[2][64][34];    // [buf][m][k]
  __shared__ __bf16 Bs[2][128][34];   // [buf][n][k]  (transposed weight tile)

  const int tid  = threadIdx.x;
  const int wid  = tid >> 5;
  const int lane = tid & 31;
  const int ln   = lane & 15;
  const int hi   = lane >> 4;
  const int wm   = wid >> 2;          // 0..1  (32-row strip)
  const int wn   = wid & 3;           // 0..3  (32-col strip)
  const int m0   = blockIdx.x * 64;
  const int n0   = blockIdx.y * 128;

  const float*  Af = (const float*)Aptr;
  const __bf16* Ab = (const __bf16*)Aptr;

  v8f acc00 = {}, acc01 = {}, acc10 = {}, acc11 = {};

  auto stage = [&](int buf, int kk) {
    // ---- A tile 64x32 : thread covers 8 consecutive k ----
    {
      int r = tid >> 2, ck = (tid & 3) * 8;
      uint32_t* d = (uint32_t*)&As[buf][r][ck];
      if (ABF16) {
        const __bf16* p = Ab + (size_t)(m0 + r) * DMODEL + kk + ck;
        uint4 v = *(const uint4*)p;
        d[0] = v.x; d[1] = v.y; d[2] = v.z; d[3] = v.w;
      } else {
        const float* p = Af + (size_t)(m0 + r) * DMODEL + kk + ck;
        __builtin_prefetch(p + 32, 0, 0);
        float4 f0 = *(const float4*)p;
        float4 f1 = *(const float4*)(p + 4);
        d[0] = bfpack(f0.x, f0.y); d[1] = bfpack(f0.z, f0.w);
        d[2] = bfpack(f1.x, f1.y); d[3] = bfpack(f1.z, f1.w);
      }
    }
    // ---- W tile 32x128 -> Bs[n][k], k-pairs packed per dword ----
    {
      int r2 = tid >> 4;            // k-pair 0..15 -> rows 2r2, 2r2+1
      int cn = (tid & 15) * 8;      // 8 consecutive n
      const float* p0 = W + (size_t)(kk + 2 * r2) * DMODEL + n0 + cn;
      const float* p1 = p0 + DMODEL;
      __builtin_prefetch(p0 + 32 * DMODEL, 0, 0);
      float4 a0 = *(const float4*)p0, a1 = *(const float4*)(p0 + 4);
      float4 b0 = *(const float4*)p1, b1 = *(const float4*)(p1 + 4);
      *(uint32_t*)&Bs[buf][cn + 0][2 * r2] = bfpack(a0.x, b0.x);
      *(uint32_t*)&Bs[buf][cn + 1][2 * r2] = bfpack(a0.y, b0.y);
      *(uint32_t*)&Bs[buf][cn + 2][2 * r2] = bfpack(a0.z, b0.z);
      *(uint32_t*)&Bs[buf][cn + 3][2 * r2] = bfpack(a0.w, b0.w);
      *(uint32_t*)&Bs[buf][cn + 4][2 * r2] = bfpack(a1.x, b1.x);
      *(uint32_t*)&Bs[buf][cn + 5][2 * r2] = bfpack(a1.y, b1.y);
      *(uint32_t*)&Bs[buf][cn + 6][2 * r2] = bfpack(a1.z, b1.z);
      *(uint32_t*)&Bs[buf][cn + 7][2 * r2] = bfpack(a1.w, b1.w);
    }
  };

  stage(0, 0);
  __syncthreads();

  const int NK = DMODEL / 32;
  for (int ks = 0; ks < NK; ++ks) {
    int buf = ks & 1;
    if (ks + 1 < NK) stage(buf ^ 1, (ks + 1) * 32);

    // ---- load all 4 fragments before the WMMA chain (overlap DS loads) ----
    v16bf a0, a1;
    #pragma unroll
    for (int v = 0; v < 8; ++v) {
      int k = ((v & 3) * 2) + hi * 8 + ((v >> 2) * 16);
      a0[2 * v]     = As[buf][wm * 32 + ln][k];
      a0[2 * v + 1] = As[buf][wm * 32 + ln][k + 1];
      a1[2 * v]     = As[buf][wm * 32 + 16 + ln][k];
      a1[2 * v + 1] = As[buf][wm * 32 + 16 + ln][k + 1];
    }
    v16bf b0, b1;
    #pragma unroll
    for (int v = 0; v < 8; ++v) {
      int k = ((v & 3) * 2) + hi * 8 + ((v >> 2) * 16);
      b0[2 * v]     = Bs[buf][wn * 32 + ln][k];
      b0[2 * v + 1] = Bs[buf][wn * 32 + ln][k + 1];
      b1[2 * v]     = Bs[buf][wn * 32 + 16 + ln][k];
      b1[2 * v + 1] = Bs[buf][wn * 32 + 16 + ln][k + 1];
    }
    acc00 = __builtin_amdgcn_wmma_f32_16x16x32_bf16(false, a0, false, b0, (short)0, acc00, false, false);
    acc01 = __builtin_amdgcn_wmma_f32_16x16x32_bf16(false, a0, false, b1, (short)0, acc01, false, false);
    acc10 = __builtin_amdgcn_wmma_f32_16x16x32_bf16(false, a1, false, b0, (short)0, acc10, false, false);
    acc11 = __builtin_amdgcn_wmma_f32_16x16x32_bf16(false, a1, false, b1, (short)0, acc11, false, false);
    __syncthreads();
  }

  // ---- epilogue: C layout is VGPR r -> row (r + 8*hi), lane ln -> col ----
  #pragma unroll
  for (int ai = 0; ai < 2; ++ai) {
    #pragma unroll
    for (int bi = 0; bi < 2; ++bi) {
      const v8f& acc = (ai == 0) ? (bi == 0 ? acc00 : acc01)
                                 : (bi == 0 ? acc10 : acc11);
      #pragma unroll
      for (int r = 0; r < 8; ++r) {
        int grow = m0 + wm * 32 + ai * 16 + r + hi * 8;
        int gc   = n0 + wn * 32 + bi * 16 + ln;
        float v  = acc[r] + bias[gc];
        if (MODE == 0) {
          ((float*)outp)[(size_t)grow * DMODEL + gc] = v;
        } else {
          int bb = grow >> 10, l = grow & 1023;
          int hh = gc >> 6,   dd = gc & 63;
          ((__bf16*)outp)[(((size_t)(bb * NHEAD + hh) * SEQ + l) * DKH) + dd] = f2bf(v);
        }
      }
    }
  }
}

// ---------------------------------------------------------------------------
// Flash attention over 3 branches (sum of per-branch softmax(QK^T/8)V).
// Block: (b,h) x 128 q-rows; 8 waves, wave owns 16 q-rows.
// K tile staged by the Tensor Data Mover (TENSORcnt); V tile transposed
// manually; both double-buffered.
// ---------------------------------------------------------------------------
__global__ __launch_bounds__(256)
void attn_kernel(const void* __restrict__ qhp,
                 const void* __restrict__ kh0p, const void* __restrict__ kh1p, const void* __restrict__ kh2p,
                 const void* __restrict__ vh0p, const void* __restrict__ vh1p, const void* __restrict__ vh2p,
                 const int* __restrict__ mask, const int* __restrict__ image_bool,
                 void* __restrict__ outcp)
{
  __shared__ __bf16 Ks[2][64][66];     // [buf][k_row][d]  (B-frag for Q@K^T)
  __shared__ __bf16 Vs[2][64][66];     // [buf][d][k_row]  transposed (B-frag for P@V)
  __shared__ __bf16 Ps[8][16][66];     // per-wave P strip (C-layout -> A-layout hop)

  const __bf16* qh = (const __bf16*)qhp;
  const __bf16* khs[3] = {(const __bf16*)kh0p, (const __bf16*)kh1p, (const __bf16*)kh2p};
  const __bf16* vhs[3] = {(const __bf16*)vh0p, (const __bf16*)vh1p, (const __bf16*)vh2p};
  __bf16* outc = (__bf16*)outcp;

  const int tid  = threadIdx.x, wid = tid >> 5, lane = tid & 31;
  const int ln   = lane & 15, hi = lane >> 4;
  const int bh   = blockIdx.y;
  const int b    = bh >> 4, h = bh & 15;
  const int q0   = blockIdx.x * 128;
  const int nBranch = (*image_bool) ? 3 : 1;

  // Q A-fragments for both 32-deep K-steps, kept in registers
  v16bf qf[2];
  {
    const __bf16* qp = qh + (((size_t)bh * SEQ) + (q0 + wid * 16 + ln)) * DKH;
    #pragma unroll
    for (int ks = 0; ks < 2; ++ks)
      #pragma unroll
      for (int v = 0; v < 8; ++v) {
        int d = ks * 32 + (v & 3) * 2 + hi * 8 + ((v >> 2) * 16);
        qf[ks][2 * v]     = qp[d];
        qf[ks][2 * v + 1] = qp[d + 1];
      }
  }

  v8f Ot0 = {}, Ot1 = {}, Ot2 = {}, Ot3 = {};

  for (int br = 0; br < nBranch; ++br) {
    const __bf16* K = khs[br] + (size_t)bh * SEQ * DKH;
    const __bf16* V = vhs[br] + (size_t)bh * SEQ * DKH;

    auto stageK = [&](int buf, int kt) {
#if USE_TDM
      if (wid == 0)
        tdm_load_tile64x64(K + (size_t)(kt * 64) * DKH,
                           (uint32_t)(uintptr_t)&Ks[buf][0][0]);
#else
      int r = tid >> 2, c = (tid & 3) * 16;
      const __bf16* kp = K + (size_t)(kt * 64 + r) * DKH + c;
      uint4 k0 = *(const uint4*)kp;
      uint4 k1 = *(const uint4*)(kp + 8);
      uint32_t* kd = (uint32_t*)&Ks[buf][r][c];
      kd[0] = k0.x; kd[1] = k0.y; kd[2] = k0.z; kd[3] = k0.w;
      kd[4] = k1.x; kd[5] = k1.y; kd[6] = k1.z; kd[7] = k1.w;
#endif
    };
    auto stageV = [&](int buf, int kt) {
      int r = tid >> 2, c = (tid & 3) * 16;
      const __bf16* vp = V + (size_t)(kt * 64 + r) * DKH + c;
      #pragma unroll
      for (int j = 0; j < 16; ++j) Vs[buf][c + j][r] = vp[j];
    };

    float mrow[8], srow[8];
    #pragma unroll
    for (int r = 0; r < 8; ++r) { mrow[r] = -1e30f; srow[r] = 0.f; }
    v8f O0 = {}, O1 = {}, O2 = {}, O3 = {};

    stageK(0, 0);
    stageV(0, 0);
#if USE_TDM
    if (wid == 0) __builtin_amdgcn_s_wait_tensorcnt(0);
#endif
    __syncthreads();

    const int NT = SEQ / 64;
    for (int kt = 0; kt < NT; ++kt) {
      int buf = kt & 1;
      if (kt + 1 < NT) { stageK(buf ^ 1, kt + 1); stageV(buf ^ 1, kt + 1); }

      // ---- S = Q @ K^T : load 4 fragments per K-step, then 4 WMMAs ----
      v8f S0 = {}, S1 = {}, S2 = {}, S3 = {};
      #pragma unroll
      for (int ksd = 0; ksd < 2; ++ksd) {
        v16bf kb0, kb1, kb2, kb3;
        #pragma unroll
        for (int v = 0; v < 8; ++v) {
          int d = ksd * 32 + (v & 3) * 2 + hi * 8 + ((v >> 2) * 16);
          kb0[2 * v] = Ks[buf][ 0 + ln][d]; kb0[2 * v + 1] = Ks[buf][ 0 + ln][d + 1];
          kb1[2 * v] = Ks[buf][16 + ln][d]; kb1[2 * v + 1] = Ks[buf][16 + ln][d + 1];
          kb2[2 * v] = Ks[buf][32 + ln][d]; kb2[2 * v + 1] = Ks[buf][32 + ln][d + 1];
          kb3[2 * v] = Ks[buf][48 + ln][d]; kb3[2 * v + 1] = Ks[buf][48 + ln][d + 1];
        }
        S0 = __builtin_amdgcn_wmma_f32_16x16x32_bf16(false, qf[ksd], false, kb0, (short)0, S0, false, false);
        S1 = __builtin_amdgcn_wmma_f32_16x16x32_bf16(false, qf[ksd], false, kb1, (short)0, S1, false, false);
        S2 = __builtin_amdgcn_wmma_f32_16x16x32_bf16(false, qf[ksd], false, kb2, (short)0, S2, false, false);
        S3 = __builtin_amdgcn_wmma_f32_16x16x32_bf16(false, qf[ksd], false, kb3, (short)0, S3, false, false);
      }

      // ---- scale, mask, online softmax, stage P (C-layout rows match reg r) ----
      #pragma unroll
      for (int r = 0; r < 8; ++r) {
        int qr = q0 + wid * 16 + r + hi * 8;
        const int* mp = mask + ((size_t)b * SEQ + qr) * SEQ + kt * 64 + ln;
        float pv[4];
        float mx = -1e30f;
        #pragma unroll
        for (int t = 0; t < 4; ++t) {
          float v = ((t == 0) ? S0[r] : (t == 1) ? S1[r] : (t == 2) ? S2[r] : S3[r]) * 0.125f;
          if (mp[t * 16] == 0) v = -1e9f;
          pv[t] = v;
          mx = fmaxf(mx, v);
        }
        #pragma unroll
        for (int o = 1; o < 16; o <<= 1) mx = fmaxf(mx, __shfl_xor(mx, o, 32));
        float mnew  = fmaxf(mrow[r], mx);
        float alpha = __expf(mrow[r] - mnew);
        float sum = 0.f;
        #pragma unroll
        for (int t = 0; t < 4; ++t) {
          float p = __expf(pv[t] - mnew);
          sum += p;
          Ps[wid][r + hi * 8][t * 16 + ln] = f2bf(p);
        }
        #pragma unroll
        for (int o = 1; o < 16; o <<= 1) sum += __shfl_xor(sum, o, 32);
        srow[r] = srow[r] * alpha + sum;
        mrow[r] = mnew;
        O0[r] *= alpha; O1[r] *= alpha; O2[r] *= alpha; O3[r] *= alpha;
      }

      // ---- O += P @ V : load 1 A-frag + 4 B-frags per K-step, then 4 WMMAs --
      #pragma unroll
      for (int ksd = 0; ksd < 2; ++ksd) {
        v16bf pa;
        #pragma unroll
        for (int v = 0; v < 8; ++v) {
          int k = ksd * 32 + (v & 3) * 2 + hi * 8 + ((v >> 2) * 16);
          pa[2 * v]     = Ps[wid][ln][k];
          pa[2 * v + 1] = Ps[wid][ln][k + 1];
        }
        v16bf vb0, vb1, vb2, vb3;
        #pragma unroll
        for (int v = 0; v < 8; ++v) {
          int k = ksd * 32 + (v & 3) * 2 + hi * 8 + ((v >> 2) * 16);
          vb0[2 * v] = Vs[buf][ 0 + ln][k]; vb0[2 * v + 1] = Vs[buf][ 0 + ln][k + 1];
          vb1[2 * v] = Vs[buf][16 + ln][k]; vb1[2 * v + 1] = Vs[buf][16 + ln][k + 1];
          vb2[2 * v] = Vs[buf][32 + ln][k]; vb2[2 * v + 1] = Vs[buf][32 + ln][k + 1];
          vb3[2 * v] = Vs[buf][48 + ln][k]; vb3[2 * v + 1] = Vs[buf][48 + ln][k + 1];
        }
        O0 = __builtin_amdgcn_wmma_f32_16x16x32_bf16(false, pa, false, vb0, (short)0, O0, false, false);
        O1 = __builtin_amdgcn_wmma_f32_16x16x32_bf16(false, pa, false, vb1, (short)0, O1, false, false);
        O2 = __builtin_amdgcn_wmma_f32_16x16x32_bf16(false, pa, false, vb2, (short)0, O2, false, false);
        O3 = __builtin_amdgcn_wmma_f32_16x16x32_bf16(false, pa, false, vb3, (short)0, O3, false, false);
      }
#if USE_TDM
      if (wid == 0) __builtin_amdgcn_s_wait_tensorcnt(0);
#endif
      __syncthreads();
    }

    #pragma unroll
    for (int r = 0; r < 8; ++r) {
      float inv = 1.0f / srow[r];
      Ot0[r] += O0[r] * inv; Ot1[r] += O1[r] * inv;
      Ot2[r] += O2[r] * inv; Ot3[r] += O3[r] * inv;
    }
  }

  // ---- write concat [b][l][h*64+d] as bf16 ----
  #pragma unroll
  for (int r = 0; r < 8; ++r) {
    int l = q0 + wid * 16 + r + hi * 8;
    size_t base = ((size_t)b * SEQ + l) * DMODEL + (size_t)h * DKH;
    outc[base +  0 + ln] = f2bf(Ot0[r]);
    outc[base + 16 + ln] = f2bf(Ot1[r]);
    outc[base + 32 + ln] = f2bf(Ot2[r]);
    outc[base + 48 + ln] = f2bf(Ot3[r]);
  }
}

// ---------------------------------------------------------------------------
extern "C" void kernel_launch(void* const* d_in, const int* in_sizes, int n_in,
                              void* d_out, int out_size, void* d_ws, size_t ws_size,
                              hipStream_t stream) {
  const float* q    = (const float*)d_in[0];
  const float* k_e  = (const float*)d_in[1];
  const float* k_i  = (const float*)d_in[2];
  const float* k_ei = (const float*)d_in[3];
  const float* v_e  = (const float*)d_in[4];
  const float* v_i  = (const float*)d_in[5];
  const float* v_ei = (const float*)d_in[6];
  const int*   mask = (const int*)d_in[7];
  const int*   image_bool = (const int*)d_in[8];
  const float* Wq   = (const float*)d_in[9];   const float* bq   = (const float*)d_in[10];
  const float* Wke  = (const float*)d_in[11];  const float* bke  = (const float*)d_in[12];
  const float* Wve  = (const float*)d_in[13];  const float* bve  = (const float*)d_in[14];
  const float* Wki  = (const float*)d_in[15];  const float* bki  = (const float*)d_in[16];
  const float* Wvi  = (const float*)d_in[17];  const float* bvi  = (const float*)d_in[18];
  const float* Wkei = (const float*)d_in[19];  const float* bkei = (const float*)d_in[20];
  const float* Wvei = (const float*)d_in[21];  const float* bvei = (const float*)d_in[22];
  const float* Wo   = (const float*)d_in[23];  const float* bo   = (const float*)d_in[24];

  const size_t SZB = (size_t)MROWS * DMODEL * sizeof(uint16_t);  // 8 MB per tensor
  char* ws  = (char*)d_ws;
  void* qh  = (void*)(ws + 0 * SZB);
  void* kh0 = (void*)(ws + 1 * SZB);
  void* kh1 = (void*)(ws + 2 * SZB);
  void* kh2 = (void*)(ws + 3 * SZB);
  void* vh0 = (void*)(ws + 4 * SZB);
  void* vh1 = (void*)(ws + 5 * SZB);
  void* vh2 = (void*)(ws + 6 * SZB);
  void* cc  = (void*)(ws + 7 * SZB);

  dim3 gg(MROWS / 64, DMODEL / 128);   // 64 x 8
  dim3 gb(256);

  // projections -> bf16 split-head layout (fp32 A, converted while staging)
  gemm_bias_kernel<0, 1><<<gg, gb, 0, stream>>>(q,    Wq,   bq,   qh);
  gemm_bias_kernel<0, 1><<<gg, gb, 0, stream>>>(k_e,  Wke,  bke,  kh0);
  gemm_bias_kernel<0, 1><<<gg, gb, 0, stream>>>(v_e,  Wve,  bve,  vh0);
  gemm_bias_kernel<0, 1><<<gg, gb, 0, stream>>>(k_i,  Wki,  bki,  kh1);
  gemm_bias_kernel<0, 1><<<gg, gb, 0, stream>>>(v_i,  Wvi,  bvi,  vh1);
  gemm_bias_kernel<0, 1><<<gg, gb, 0, stream>>>(k_ei, Wkei, bkei, kh2);
  gemm_bias_kernel<0, 1><<<gg, gb, 0, stream>>>(v_ei, Wvei, bvei, vh2);

  // fused 3-branch flash attention -> bf16 concat
  dim3 ag(SEQ / 128, BATCH * NHEAD);  // 8 x 64
  attn_kernel<<<ag, gb, 0, stream>>>(qh, kh0, kh1, kh2, vh0, vh1, vh2, mask, image_bool, cc);

  // output projection: bf16 A from workspace -> fp32 d_out
  gemm_bias_kernel<1, 0><<<gg, gb, 0, stream>>>(cc, Wo, bo, d_out);
}
